// MaskedMSA_45466523795480
// MI455X (gfx1250) — matrix operand order repaired and checked by reading, not compiled, not run on hardware
//
#include <hip/hip_runtime.h>

// ---------------------------------------------------------------------------
// MaskedMSA for MI455X (gfx1250): pack -> QKV GEMM -> causal flash attention
// -> out GEMM. Matmuls on v_wmma_f32_16x16x32_f16 (wave32). GEMM tiles staged
// by the Tensor Data Mover (tensor_load_to_lds + s_wait_tensorcnt, double
// buffered); attention K tiles staged by GLOBAL_LOAD_ASYNC_TO_LDS_B128
// (ASYNCcnt); softmax reductions via ds_swizzle.
// ---------------------------------------------------------------------------

typedef __attribute__((ext_vector_type(4)))  _Float16 h4;
typedef __attribute__((ext_vector_type(8)))  _Float16 h8;
typedef __attribute__((ext_vector_type(16))) _Float16 h16;
typedef __attribute__((ext_vector_type(8)))  float    f8;
typedef __attribute__((ext_vector_type(4)))  unsigned int u32x4;
typedef __attribute__((ext_vector_type(8)))  int i32x8;
typedef __attribute__((ext_vector_type(4)))  int i32x4;

#define NB   4
#define SEQ  2048
#define EMB  1024
#define HID  1024
#define NH   16
#define QKVN 3072          // 3*HID
#define MTOT 8192          // NB*SEQ

#if defined(__has_builtin)
#if __has_builtin(__builtin_amdgcn_tensor_load_to_lds)
#define HAVE_TDM 1
#endif
#endif
#ifndef HAVE_TDM
#define HAVE_TDM 0
#endif

// 16-lane xor butterfly via DS_SWIZZLE_B32 (group-of-32: offset = xor<<10 | 0x1f)
#define SWZ_XOR_F(v, m) \
  __int_as_float(__builtin_amdgcn_ds_swizzle(__float_as_int(v), ((m) << 10) | 0x1f))

__device__ __forceinline__ float redmax16(float v) {
  v = fmaxf(v, SWZ_XOR_F(v, 8));
  v = fmaxf(v, SWZ_XOR_F(v, 4));
  v = fmaxf(v, SWZ_XOR_F(v, 2));
  v = fmaxf(v, SWZ_XOR_F(v, 1));
  return v;
}
__device__ __forceinline__ float redsum16(float v) {
  v += SWZ_XOR_F(v, 8);
  v += SWZ_XOR_F(v, 4);
  v += SWZ_XOR_F(v, 2);
  v += SWZ_XOR_F(v, 1);
  return v;
}

// Per-lane async DMA: LDS[lds_off..+15] = MEM[base + gl_off .. +15]  (ASYNCcnt)
__device__ __forceinline__ void async_b128(unsigned lds_off, unsigned gl_off,
                                           const void* base) {
  asm volatile("global_load_async_to_lds_b128 %0, %1, %2"
               :
               : "v"(lds_off), "v"(gl_off),
                 "s"((unsigned long long)(uintptr_t)base)
               : "memory");
}
#define WAIT_ASYNC(n) asm volatile("s_wait_asynccnt " #n ::: "memory")

#if HAVE_TDM
// TDM: DMA one 2D f16 tile (tile_d0 x tile_d1 elements, row stride = stride_d0
// elements) from global into LDS, inserting 16B of LDS padding after every
// 64B row chunk (pad_interval=16 DW, pad_amount=4 DW) -> 40-half LDS rows.
__device__ __forceinline__ void tdm_load_2d_f16(
    unsigned lds_addr, const void* gaddr, unsigned tensor_d0, unsigned tensor_d1,
    unsigned tile_d0, unsigned tile_d1, unsigned stride_d0) {
  unsigned long long ga = (unsigned long long)(uintptr_t)gaddr;
  u32x4 g0;
  g0[0] = 1u;                                        // count=1, user mode
  g0[1] = lds_addr;                                  // lds_addr [63:32]
  g0[2] = (unsigned)(ga & 0xFFFFFFFFu);              // global_addr lo
  g0[3] = (unsigned)((ga >> 32) & 0x01FFFFFFu) | 0x80000000u;  // hi25 | type=2
  i32x8 g1;
  g1[0] = (int)((1u << 16) | (1u << 20) | (3u << 22) | (3u << 25));
  //        data_size=2B   pad_en       pad_int=64B   pad_amt=16B
  g1[1] = (int)((tensor_d0 & 0xFFFFu) << 16);
  g1[2] = (int)(((tensor_d0 >> 16) & 0xFFFFu) | ((tensor_d1 & 0xFFFFu) << 16));
  g1[3] = (int)(((tensor_d1 >> 16) & 0xFFFFu) | ((tile_d0 & 0xFFFFu) << 16));
  g1[4] = (int)(tile_d1 & 0xFFFFu);                  // tile_dim1; tile_dim2=0
  g1[5] = (int)stride_d0;                            // dim0_stride lo32
  g1[6] = 0;
  g1[7] = 0;
  i32x4 z4 = {0, 0, 0, 0};
  i32x8 z8 = {0, 0, 0, 0, 0, 0, 0, 0};
  __builtin_amdgcn_tensor_load_to_lds(g0, g1, z4, z4, z8, 0);
}
#endif

__device__ __forceinline__ h16 cat8(h8 lo, h8 hi) {
  h16 r;
#pragma unroll
  for (int i = 0; i < 8; ++i) { r[i] = lo[i]; r[i + 8] = hi[i]; }
  return r;
}

__device__ __forceinline__ f8 wmma16(h16 a, h16 b, f8 c) {
  return __builtin_amdgcn_wmma_f32_16x16x32_f16(false, a, false, b, (short)0, c,
                                                false, false);
}

// ---------------------------------------------------------------------------
// Pre-pass A: fp32 -> f16 pack (vectorized x4)
// ---------------------------------------------------------------------------
__global__ __launch_bounds__(256) void pack_half(const float* __restrict__ src,
                                                 _Float16* __restrict__ dst) {
  int i = blockIdx.x * 256 + threadIdx.x;
  float4 f = ((const float4*)src)[i];
  h4 h = {(_Float16)f.x, (_Float16)f.y, (_Float16)f.z, (_Float16)f.w};
  ((h4*)dst)[i] = h;
}

// ---------------------------------------------------------------------------
// Pre-pass B: fp32 W[K][N] -> f16 Wt[N][K] (32x32 LDS tile transpose)
// ---------------------------------------------------------------------------
__global__ __launch_bounds__(256) void pack_transpose(
    const float* __restrict__ W, _Float16* __restrict__ Wt, int K, int N) {
  __shared__ float T[32][33];
  const int k0 = blockIdx.y * 32, n0 = blockIdx.x * 32;
  const int tid = threadIdx.x;
#pragma unroll
  for (int ii = 0; ii < 4; ++ii) {
    int lin = ii * 256 + tid, kl = lin >> 5, nl = lin & 31;
    T[kl][nl] = W[(size_t)(k0 + kl) * N + n0 + nl];
  }
  __syncthreads();
#pragma unroll
  for (int ii = 0; ii < 4; ++ii) {
    int lin = ii * 256 + tid, nl = lin >> 5, kl = lin & 31;
    Wt[(size_t)(n0 + nl) * K + k0 + kl] = (_Float16)T[kl][nl];
  }
}

// ---------------------------------------------------------------------------
// f16 GEMM + bias: Out[M,N] = A[M,K] * Bt[N,K]^T + bias[N]
// BM=BN=128, BK=32; 8 waves in 4x2, 32x64 per wave (2x4 WMMA tiles).
// Double-buffered LDS; tiles staged by ONE tensor_load_to_lds each (wave 0
// issues, TDM ops complete in order per wave so tensorcnt<=2 => previous
// A+B pair landed). Fallback: per-lane async b128 copies.
// ---------------------------------------------------------------------------
template <bool OUTHALF>
__global__ __launch_bounds__(256) void gemm_f16_wmma(
    const _Float16* __restrict__ A, const _Float16* __restrict__ Bt,
    const float* __restrict__ bias, void* __restrict__ Ov, int M, int N,
    int K) {
  __shared__ _Float16 As[2][128 * 40];   // [row][k], stride 40 halves
  __shared__ _Float16 Bs[2][128 * 40];   // [n][k]

  const int tid  = threadIdx.x;
  const int wave = tid >> 5, lane = tid & 31;
  const int wr = wave >> 1, wc = wave & 1;
  const int r16 = lane & 15, hs = lane >> 4;
  const int mb = blockIdx.y, nb = blockIdx.x;

#if !HAVE_TDM
  const int sr0 = tid >> 2, sc0 = (tid & 3) * 8;
  const int sr1 = (256 + tid) >> 2, sc1 = sc0;
#endif

  f8 acc[2][4];
#pragma unroll
  for (int i = 0; i < 2; ++i)
#pragma unroll
    for (int j = 0; j < 4; ++j) acc[i][j] = (f8){};

  auto stage = [&](int k0, int buf) {
#if HAVE_TDM
    if (wave == 0) {
      tdm_load_2d_f16((unsigned)(uintptr_t)&As[buf][0],
                      A + (size_t)(mb * 128) * K + k0, K, M, 32, 128, K);
      tdm_load_2d_f16((unsigned)(uintptr_t)&Bs[buf][0],
                      Bt + (size_t)(nb * 128) * K + k0, K, N, 32, 128, K);
    }
#else
    async_b128((unsigned)(uintptr_t)&As[buf][sr0 * 40 + sc0],
               (unsigned)(((size_t)(mb * 128 + sr0) * K + k0 + sc0) * 2), A);
    async_b128((unsigned)(uintptr_t)&As[buf][sr1 * 40 + sc1],
               (unsigned)(((size_t)(mb * 128 + sr1) * K + k0 + sc1) * 2), A);
    async_b128((unsigned)(uintptr_t)&Bs[buf][sr0 * 40 + sc0],
               (unsigned)(((size_t)(nb * 128 + sr0) * K + k0 + sc0) * 2), Bt);
    async_b128((unsigned)(uintptr_t)&Bs[buf][sr1 * 40 + sc1],
               (unsigned)(((size_t)(nb * 128 + sr1) * K + k0 + sc1) * 2), Bt);
#endif
  };

  stage(0, 0);
  int cur = 0;
  for (int k0 = 0; k0 < K; k0 += 32) {
    __syncthreads();                       // readers of buf cur^1 are done
    const bool more = (k0 + 32) < K;
    if (more) stage(k0 + 32, cur ^ 1);     // prefetch next tile
#if HAVE_TDM
    if (wave == 0) {
      if (more) __builtin_amdgcn_s_wait_tensorcnt(2);
      else      __builtin_amdgcn_s_wait_tensorcnt(0);
    }
#else
    if (more) WAIT_ASYNC(0x4);
    else      WAIT_ASYNC(0x0);
#endif
    __syncthreads();                       // everyone sees the current tile

    // A fragments (16x32): lanes 0-15 hold K 0-7 & 16-23; 16-31: 8-15 & 24-31
    h16 afr[2];
#pragma unroll
    for (int mi = 0; mi < 2; ++mi) {
      const _Float16* ap = &As[cur][(wr * 32 + mi * 16 + r16) * 40 + hs * 8];
      afr[mi] = cat8(*(const h8*)ap, *(const h8*)(ap + 16));
    }
    // B fragments (32x16): lanes 0-15 hold K 0-7 & 16-23; 16-31: 8-15 & 24-31
    h16 bfr[4];
#pragma unroll
    for (int ni = 0; ni < 4; ++ni) {
      const _Float16* bp = &Bs[cur][(wc * 64 + ni * 16 + r16) * 40 + hs * 8];
      bfr[ni] = cat8(*(const h8*)bp, *(const h8*)(bp + 16));
    }
#pragma unroll
    for (int mi = 0; mi < 2; ++mi)
#pragma unroll
      for (int ni = 0; ni < 4; ++ni)
        acc[mi][ni] = wmma16(afr[mi], bfr[ni], acc[mi][ni]);
    cur ^= 1;
  }

  // epilogue: C layout VGPR g = row (g + hs*8), lane r16 = col
#pragma unroll
  for (int mi = 0; mi < 2; ++mi)
#pragma unroll
    for (int ni = 0; ni < 4; ++ni)
#pragma unroll
      for (int g = 0; g < 8; ++g) {
        int row = mb * 128 + wr * 32 + mi * 16 + g + hs * 8;
        int col = nb * 128 + wc * 64 + ni * 16 + r16;
        float v = acc[mi][ni][g] + bias[col];
        if (OUTHALF)
          ((_Float16*)Ov)[(size_t)row * N + col] = (_Float16)v;
        else
          ((float*)Ov)[(size_t)row * N + col] = v;
      }
}

// ---------------------------------------------------------------------------
// Causal flash attention. Under the reference's mixed reshape the per-(b,h)
// views are UNIFORMLY strided:
//   Q/K/V[t', e] = QKV[(b*SEQ + 128h)*3072 + 192*t' + {0,64,128} + e]
//   SA  [s', e] -> (b*SEQ + 128h)*1024 + 64*s' + e
// grid = (16 row-blocks, 64 (b,h)), 8 waves x 16 s'-rows, key tiles of 32.
// ---------------------------------------------------------------------------
__global__ __launch_bounds__(256) void attn_flash_wmma(
    const _Float16* __restrict__ QKV, _Float16* __restrict__ SA) {
  __shared__ _Float16 Kt[32][72];       // K tile [t][e] (async-staged)
  __shared__ _Float16 Vt[64][40];       // V tile transposed [e][t]
  __shared__ _Float16 Pst[8][16][40];   // per-wave P staging [row][t]

  const int ib = blockIdx.x;
  const int bh = blockIdx.y;
  const int b = bh >> 4, h = bh & 15;
  const int tid = threadIdx.x;
  const int wave = tid >> 5, lane = tid & 31;
  const int r16 = lane & 15, hs = lane >> 4;
  const float scale = 0.022097086912079608f;   // (2*HID)^-0.5

  const size_t qkv0 = (size_t)(b * SEQ + h * 128) * QKVN;   // head base
  const size_t sa0  = (size_t)(b * SEQ + h * 128) * HID;

  // Q fragments, loaded once directly from global in A-layout
  const int srow = ib * 128 + wave * 16 + r16;
  h16 aQ[2];
#pragma unroll
  for (int ef = 0; ef < 2; ++ef) {
    const _Float16* qp = QKV + qkv0 + (size_t)192 * srow + ef * 32 + hs * 8;
    aQ[ef] = cat8(*(const h8*)qp, *(const h8*)(qp + 16));
  }

  f8 o[4];
#pragma unroll
  for (int eb = 0; eb < 4; ++eb) o[eb] = (f8){};
  float mrow[8], lrow[8];
#pragma unroll
  for (int g = 0; g < 8; ++g) { mrow[g] = -1e30f; lrow[g] = 0.0f; }

  // staging coords: 32 t-rows x 8 e-chunks of 8 halves
  const int strow = tid >> 3;
  const int se0 = (tid & 7) * 8;

  const int jmax = 4 * ib + 3;
  for (int j = 0; j <= jmax; ++j) {
    __syncthreads();
    {
      size_t base = qkv0 + (size_t)192 * (j * 32 + strow);
      // K tile: straight strided copy -> async DMA (1 instr/wave, 16B/lane)
      async_b128((unsigned)(uintptr_t)&Kt[strow][se0],
                 (unsigned)((base + 64 + se0) * 2), QKV);
      // V tile: needs transpose -> manual scatter
      h8 vv = *(const h8*)(QKV + base + 128 + se0);
#pragma unroll
      for (int q = 0; q < 8; ++q) Vt[se0 + q][strow] = vv[q];
    }
    WAIT_ASYNC(0x0);
    __syncthreads();

    // S = Q * K^T : 2 col-tiles, e-dim 64 -> 2 WMMAs each
    f8 s[2];
#pragma unroll
    for (int ct = 0; ct < 2; ++ct) {
      f8 sc = (f8){};
#pragma unroll
      for (int ef = 0; ef < 2; ++ef) {
        const _Float16* kp = &Kt[ct * 16 + r16][ef * 32 + hs * 8];
        h16 bk = cat8(*(const h8*)kp, *(const h8*)(kp + 16));
        sc = wmma16(aQ[ef], bk, sc);
      }
      s[ct] = sc;
    }

    // scale + causal mask (t' <= s')
#pragma unroll
    for (int ct = 0; ct < 2; ++ct)
#pragma unroll
      for (int g = 0; g < 8; ++g) {
        int sg = ib * 128 + wave * 16 + g + hs * 8;
        int tg = j * 32 + ct * 16 + r16;
        float v = s[ct][g] * scale;
        s[ct][g] = (tg <= sg) ? v : -1e30f;
      }

    // online softmax: row lives in one VGPR across a 16-lane group
    float corr[8];
#pragma unroll
    for (int g = 0; g < 8; ++g) {
      float v = redmax16(fmaxf(s[0][g], s[1][g]));
      float mnew = fmaxf(mrow[g], v);
      corr[g] = __expf(mrow[g] - mnew);
      mrow[g] = mnew;
    }
#pragma unroll
    for (int ct = 0; ct < 2; ++ct)
#pragma unroll
      for (int g = 0; g < 8; ++g) s[ct][g] = __expf(s[ct][g] - mrow[g]);
#pragma unroll
    for (int g = 0; g < 8; ++g) {
      float rs = redsum16(s[0][g] + s[1][g]);
      lrow[g] = lrow[g] * corr[g] + rs;
    }
#pragma unroll
    for (int eb = 0; eb < 4; ++eb)
#pragma unroll
      for (int g = 0; g < 8; ++g) o[eb][g] *= corr[g];

    // P: C-layout -> A-fragment layout via per-wave LDS patch
#pragma unroll
    for (int ct = 0; ct < 2; ++ct)
#pragma unroll
      for (int g = 0; g < 8; ++g)
        Pst[wave][g + hs * 8][ct * 16 + r16] = (_Float16)s[ct][g];
    __syncthreads();

    const _Float16* pp = &Pst[wave][r16][hs * 8];
    h16 aP = cat8(*(const h8*)pp, *(const h8*)(pp + 16));

    // O += P * V  (4 e-blocks)
#pragma unroll
    for (int eb = 0; eb < 4; ++eb) {
      const _Float16* vp = &Vt[eb * 16 + r16][hs * 8];
      h16 bv = cat8(*(const h8*)vp, *(const h8*)(vp + 16));
      o[eb] = wmma16(aP, bv, o[eb]);
    }
  }

  // normalize + store (per-head stride-64 layout)
#pragma unroll
  for (int eb = 0; eb < 4; ++eb)
#pragma unroll
    for (int g = 0; g < 8; ++g) {
      int sg = ib * 128 + wave * 16 + g + hs * 8;
      float v = o[eb][g] / lrow[g];
      SA[sa0 + (size_t)64 * sg + eb * 16 + r16] = (_Float16)v;
    }
}

// ---------------------------------------------------------------------------
extern "C" void kernel_launch(void* const* d_in, const int* in_sizes, int n_in,
                              void* d_out, int out_size, void* d_ws,
                              size_t ws_size, hipStream_t stream) {
  (void)in_sizes; (void)n_in; (void)out_size; (void)ws_size;
  const float* X    = (const float*)d_in[0];   // [8192, 1024]
  const float* Wqkv = (const float*)d_in[1];   // [1024, 3072]
  const float* bqkv = (const float*)d_in[2];   // [3072]
  const float* Wout = (const float*)d_in[3];   // [1024, 1024]
  const float* bout = (const float*)d_in[4];   // [1024]
  float* out = (float*)d_out;                  // [8192, 1024] fp32

  char* ws = (char*)d_ws;
  _Float16* Xh    = (_Float16*)(ws);                              // 16 MB
  _Float16* Wqkvt = (_Float16*)(ws + (size_t)16 * 1024 * 1024);   //  6 MB
  _Float16* Wot   = (_Float16*)(ws + (size_t)22 * 1024 * 1024);   //  2 MB
  _Float16* QKV   = (_Float16*)(ws + (size_t)24 * 1024 * 1024);   // 48 MB
  _Float16* SA    = (_Float16*)(ws + (size_t)72 * 1024 * 1024);   // 16 MB

  dim3 blk(256);
  pack_half<<<dim3((size_t)MTOT * EMB / 1024), blk, 0, stream>>>(X, Xh);
  pack_transpose<<<dim3(QKVN / 32, EMB / 32), blk, 0, stream>>>(Wqkv, Wqkvt,
                                                                EMB, QKVN);
  pack_transpose<<<dim3(HID / 32, HID / 32), blk, 0, stream>>>(Wout, Wot, HID,
                                                               HID);
  dim3 g1(QKVN / 128, MTOT / 128);
  gemm_f16_wmma<true><<<g1, blk, 0, stream>>>(Xh, Wqkvt, bqkv, QKV, MTOT, QKVN,
                                              EMB);
  dim3 g2(SEQ / 128, NB * NH);
  attn_flash_wmma<<<g2, blk, 0, stream>>>(QKV, SA);
  dim3 g3(HID / 128, MTOT / 128);
  gemm_f16_wmma<false><<<g3, blk, 0, stream>>>(SA, Wot, bout, out, MTOT, HID,
                                               HID);
}
